// LlamaAttention_23519240913221
// MI455X (gfx1250) — matrix operand order repaired and checked by reading, not compiled, not run on hardware
//
#include <hip/hip_runtime.h>
#include <cstdint>

typedef __attribute__((ext_vector_type(16))) _Float16 v16h;
typedef __attribute__((ext_vector_type(8)))  _Float16 v8h;
typedef __attribute__((ext_vector_type(8)))  float    v8f;

#define D_MODEL 2048
#define T_SEQ   2048
#define NBATCH  2
#define NHEADS  16
#define HDIM    128

#define ASTRIDE 72              // padded LDS row stride (halves): 144B, 16B-aligned
#define ABUF    (32 * ASTRIDE)  // one 32x64 A tile (padded)

__device__ __forceinline__ v16h cat8(v8h lo, v8h hi) {
  v16h r;
#pragma unroll
  for (int i = 0; i < 8; ++i) { r[i] = lo[i]; r[i + 8] = hi[i]; }
  return r;
}

// ---- CDNA5 async global->LDS staging (ASYNCcnt-tracked) ---------------------
// Stage A[m0..m0+31, k0..k0+63] (f16) into a padded LDS tile. 128 threads,
// 2 x b128 chunks each. Low 32 bits of a generic pointer into __shared__ are
// the LDS byte offset (flat LDS aperture keeps the offset in addr[31:0]).
__device__ __forceinline__ void async_copy_tileA(const _Float16* __restrict__ A,
                                                 int K, int m0, int k0, int tid,
                                                 _Float16* buf) {
#pragma unroll
  for (int c = tid; c < 256; c += 128) {
    int row = c >> 3, colh = (c & 7) * 8;
    const _Float16* gsrc = A + (size_t)(m0 + row) * K + k0 + colh;
    unsigned loff = (unsigned)(uintptr_t)(buf + row * ASTRIDE + colh);
    asm volatile("global_load_async_to_lds_b128 %0, %1, off"
                 :: "v"(loff), "v"(gsrc) : "memory");
  }
}
__device__ __forceinline__ void wait_async_le2() {
  asm volatile("s_wait_asynccnt 2" ::: "memory");
}
__device__ __forceinline__ void wait_async_0() {
  asm volatile("s_wait_asynccnt 0" ::: "memory");
}

// ---------------------------------------------------------------- conversions
__global__ void cvt_f32_f16(const float* __restrict__ src,
                            _Float16* __restrict__ dst, int n) {
  int i = blockIdx.x * blockDim.x + threadIdx.x;
  if (i < n) dst[i] = (_Float16)src[i];
}

// ------------------------------------------------- C[m,n] = sum_k A[m,k]B[n,k]
// A: MxK f16 row-major, B: NxK f16 row-major. Block = 4 waves; A tile shared
// through LDS via async double-buffer; each wave owns a 32x32 output tile.
// Fragment layouts per CDNA5 ISA 7.12.2.
__global__ void gemm_nt_f16(const _Float16* __restrict__ A,
                            const _Float16* __restrict__ B,
                            _Float16* __restrict__ C, int M, int N, int K) {
  __shared__ __align__(16) _Float16 Alds[2][ABUF];
  const int lane = threadIdx.x;
  const int hl = lane >> 4, l16 = lane & 15;
  const int tid = threadIdx.y * 32 + lane;
  const int m0 = blockIdx.y * 32;
  const int n0 = (blockIdx.x * blockDim.y + threadIdx.y) * 32;
  v8f acc00 = {}, acc01 = {}, acc10 = {}, acc11 = {};
  const _Float16* Br0 = B + (size_t)(n0 + l16) * K;
  const _Float16* Br1 = B + (size_t)(n0 + 16 + l16) * K;

  async_copy_tileA(A, K, m0, 0, tid, Alds[0]);
  const int TSTEPS = K / 64;
  for (int t = 0; t < TSTEPS; ++t) {
    const int k0 = t * 64;
    if (t + 1 < TSTEPS) {
      async_copy_tileA(A, K, m0, k0 + 64, tid, Alds[(t + 1) & 1]);
      wait_async_le2();             // tile t landed; tile t+1 still in flight
    } else {
      wait_async_0();
    }
    __syncthreads();
    const _Float16* Ab = Alds[t & 1];
#pragma unroll
    for (int ks = 0; ks < 2; ++ks) {
      const int kk = ks * 32;
      v16h a0 = cat8(*(const v8h*)(Ab + l16 * ASTRIDE + kk + hl * 8),
                     *(const v8h*)(Ab + l16 * ASTRIDE + kk + 16 + hl * 8));
      v16h a1 = cat8(*(const v8h*)(Ab + (16 + l16) * ASTRIDE + kk + hl * 8),
                     *(const v8h*)(Ab + (16 + l16) * ASTRIDE + kk + 16 + hl * 8));
      v16h b0 = cat8(*(const v8h*)(Br0 + k0 + kk + hl * 8),
                     *(const v8h*)(Br0 + k0 + kk + 16 + hl * 8));
      v16h b1 = cat8(*(const v8h*)(Br1 + k0 + kk + hl * 8),
                     *(const v8h*)(Br1 + k0 + kk + 16 + hl * 8));
      acc00 = __builtin_amdgcn_wmma_f32_16x16x32_f16(false, a0, false, b0, (short)0, acc00, false, false);
      acc01 = __builtin_amdgcn_wmma_f32_16x16x32_f16(false, a0, false, b1, (short)0, acc01, false, false);
      acc10 = __builtin_amdgcn_wmma_f32_16x16x32_f16(false, a1, false, b0, (short)0, acc10, false, false);
      acc11 = __builtin_amdgcn_wmma_f32_16x16x32_f16(false, a1, false, b1, (short)0, acc11, false, false);
    }
    __syncthreads();                // buf (t&1) free before t+1 issues into it
  }
#pragma unroll
  for (int r = 0; r < 8; ++r) {
    int m = r + 8 * hl;
    C[(size_t)(m0 + m) * N + n0 + l16]           = (_Float16)acc00[r];
    C[(size_t)(m0 + m) * N + n0 + 16 + l16]      = (_Float16)acc01[r];
    C[(size_t)(m0 + 16 + m) * N + n0 + l16]      = (_Float16)acc10[r];
    C[(size_t)(m0 + 16 + m) * N + n0 + 16 + l16] = (_Float16)acc11[r];
  }
}

// same GEMM but fp32 output + bias (final projection)
__global__ void gemm_nt_f32_bias(const _Float16* __restrict__ A,
                                 const _Float16* __restrict__ B,
                                 const float* __restrict__ bias,
                                 float* __restrict__ C, int M, int N, int K) {
  __shared__ __align__(16) _Float16 Alds[2][ABUF];
  const int lane = threadIdx.x;
  const int hl = lane >> 4, l16 = lane & 15;
  const int tid = threadIdx.y * 32 + lane;
  const int m0 = blockIdx.y * 32;
  const int n0 = (blockIdx.x * blockDim.y + threadIdx.y) * 32;
  v8f acc00 = {}, acc01 = {}, acc10 = {}, acc11 = {};
  const _Float16* Br0 = B + (size_t)(n0 + l16) * K;
  const _Float16* Br1 = B + (size_t)(n0 + 16 + l16) * K;

  async_copy_tileA(A, K, m0, 0, tid, Alds[0]);
  const int TSTEPS = K / 64;
  for (int t = 0; t < TSTEPS; ++t) {
    const int k0 = t * 64;
    if (t + 1 < TSTEPS) {
      async_copy_tileA(A, K, m0, k0 + 64, tid, Alds[(t + 1) & 1]);
      wait_async_le2();
    } else {
      wait_async_0();
    }
    __syncthreads();
    const _Float16* Ab = Alds[t & 1];
#pragma unroll
    for (int ks = 0; ks < 2; ++ks) {
      const int kk = ks * 32;
      v16h a0 = cat8(*(const v8h*)(Ab + l16 * ASTRIDE + kk + hl * 8),
                     *(const v8h*)(Ab + l16 * ASTRIDE + kk + 16 + hl * 8));
      v16h a1 = cat8(*(const v8h*)(Ab + (16 + l16) * ASTRIDE + kk + hl * 8),
                     *(const v8h*)(Ab + (16 + l16) * ASTRIDE + kk + 16 + hl * 8));
      v16h b0 = cat8(*(const v8h*)(Br0 + k0 + kk + hl * 8),
                     *(const v8h*)(Br0 + k0 + kk + 16 + hl * 8));
      v16h b1 = cat8(*(const v8h*)(Br1 + k0 + kk + hl * 8),
                     *(const v8h*)(Br1 + k0 + kk + 16 + hl * 8));
      acc00 = __builtin_amdgcn_wmma_f32_16x16x32_f16(false, a0, false, b0, (short)0, acc00, false, false);
      acc01 = __builtin_amdgcn_wmma_f32_16x16x32_f16(false, a0, false, b1, (short)0, acc01, false, false);
      acc10 = __builtin_amdgcn_wmma_f32_16x16x32_f16(false, a1, false, b0, (short)0, acc10, false, false);
      acc11 = __builtin_amdgcn_wmma_f32_16x16x32_f16(false, a1, false, b1, (short)0, acc11, false, false);
    }
    __syncthreads();
  }
  float bz0 = bias[n0 + l16], bz1 = bias[n0 + 16 + l16];
#pragma unroll
  for (int r = 0; r < 8; ++r) {
    int m = r + 8 * hl;
    C[(size_t)(m0 + m) * N + n0 + l16]           = acc00[r] + bz0;
    C[(size_t)(m0 + m) * N + n0 + 16 + l16]      = acc01[r] + bz1;
    C[(size_t)(m0 + 16 + m) * N + n0 + l16]      = acc10[r] + bz0;
    C[(size_t)(m0 + 16 + m) * N + n0 + 16 + l16] = acc11[r] + bz1;
  }
}

// ------------------------------------------- RoPE + layout [b,t,h,d]->[b,h,t,d]
__global__ void rope_qk(const _Float16* __restrict__ Qt,
                        const _Float16* __restrict__ Kt,
                        const float* __restrict__ theta,
                        _Float16* __restrict__ Qh, _Float16* __restrict__ Kh) {
  int idx = blockIdx.x * blockDim.x + threadIdx.x;  // B*T*H*64
  if (idx >= NBATCH * T_SEQ * NHEADS * (HDIM / 2)) return;
  int i = idx & 63;
  int h = (idx >> 6) & (NHEADS - 1);
  int t = (idx >> 10) & (T_SEQ - 1);
  int b = idx >> 21;
  float fr = (float)t * theta[i];
  float c, s;
  __sincosf(fr, &s, &c);
  size_t src = (size_t)(b * T_SEQ + t) * D_MODEL + h * HDIM;
  size_t dst = ((size_t)(b * NHEADS + h) * T_SEQ + t) * HDIM;
  float eq = (float)Qt[src + 2 * i], oq = (float)Qt[src + 2 * i + 1];
  float ek = (float)Kt[src + 2 * i], ok = (float)Kt[src + 2 * i + 1];
  Qh[dst + i]      = (_Float16)(eq * c - oq * s);
  Qh[dst + 64 + i] = (_Float16)(eq * s + oq * c);
  Kh[dst + i]      = (_Float16)(ek * c - ok * s);
  Kh[dst + 64 + i] = (_Float16)(ek * s + ok * c);
}

// ---------------------------------------- V: [b,t,h,d] -> [b,h,d,t] (for P*V)
__global__ void v_transpose(const _Float16* __restrict__ Vtmp,
                            _Float16* __restrict__ Vt) {
  int idx = blockIdx.x * blockDim.x + threadIdx.x;  // B*T*H*128
  if (idx >= NBATCH * T_SEQ * NHEADS * HDIM) return;
  int d = idx & 127;
  int h = (idx >> 7) & (NHEADS - 1);
  int t = (idx >> 11) & (T_SEQ - 1);
  int b = idx >> 22;
  Vt[((size_t)(b * NHEADS + h) * HDIM + d) * T_SEQ + t] =
      Vtmp[(size_t)(b * T_SEQ + t) * D_MODEL + h * HDIM + d];
}

// ------------------------------------------------ flash attention, 1 wave/tile
__global__ void attn_kernel(const _Float16* __restrict__ Qh,
                            const _Float16* __restrict__ Kh,
                            const _Float16* __restrict__ Vt,
                            _Float16* __restrict__ ctx) {
  __shared__ __align__(16) _Float16 Plds[16 * 40];  // padded 16x32 P tile
  const int lane = threadIdx.x;
  const int hl = lane >> 4, l16 = lane & 15;
  const int q0 = blockIdx.x * 16;
  const int h = blockIdx.y;
  const int b = blockIdx.z;
  const _Float16* Qb = Qh + (size_t)(b * NHEADS + h) * T_SEQ * HDIM;
  const _Float16* Kb = Kh + (size_t)(b * NHEADS + h) * T_SEQ * HDIM;
  const _Float16* Vb = Vt + (size_t)(b * NHEADS + h) * HDIM * T_SEQ;

  // Q A-fragments for rows q0..q0+15, 4 d-steps of 32
  v16h qa[4];
  {
    const _Float16* qrow = Qb + (size_t)(q0 + l16) * HDIM;
#pragma unroll
    for (int ds = 0; ds < 4; ++ds)
      qa[ds] = cat8(*(const v8h*)(qrow + ds * 32 + hl * 8),
                    *(const v8h*)(qrow + ds * 32 + 16 + hl * 8));
  }
  v8f o[8] = {};
  float mrow[8], lrow[8];
#pragma unroll
  for (int r = 0; r < 8; ++r) { mrow[r] = -3.0e38f; lrow[r] = 0.0f; }
  const float scale = 0.08838834764831845f;  // 1/sqrt(128)
  const int kmax = q0 + 16;
  for (int k0 = 0; k0 < kmax; k0 += 32) {
    // S = Q K^T : two 16x16 C tiles (cols k0..k0+15, k0+16..k0+31)
    v8f c0 = {}, c1 = {};
    const _Float16* kr0 = Kb + (size_t)(k0 + l16) * HDIM;
    const _Float16* kr1 = Kb + (size_t)(k0 + 16 + l16) * HDIM;
#pragma unroll
    for (int ds = 0; ds < 4; ++ds) {
      v16h kb0 = cat8(*(const v8h*)(kr0 + ds * 32 + hl * 8),
                      *(const v8h*)(kr0 + ds * 32 + 16 + hl * 8));
      v16h kb1 = cat8(*(const v8h*)(kr1 + ds * 32 + hl * 8),
                      *(const v8h*)(kr1 + ds * 32 + 16 + hl * 8));
      c0 = __builtin_amdgcn_wmma_f32_16x16x32_f16(false, qa[ds], false, kb0, (short)0, c0, false, false);
      c1 = __builtin_amdgcn_wmma_f32_16x16x32_f16(false, qa[ds], false, kb1, (short)0, c1, false, false);
    }
    // online softmax (rows live in 16-lane halves; width-16 reductions)
#pragma unroll
    for (int r = 0; r < 8; ++r) {
      int m = r + 8 * hl;
      int q = q0 + m;
      float x0 = c0[r] * scale;
      float x1 = c1[r] * scale;
      if (k0 + l16 > q)      x0 = -3.0e38f;
      if (k0 + 16 + l16 > q) x1 = -3.0e38f;
      float mx = fmaxf(x0, x1);
#pragma unroll
      for (int off = 1; off < 16; off <<= 1)
        mx = fmaxf(mx, __shfl_xor(mx, off, 16));
      float newm = fmaxf(mrow[r], mx);
      float alpha = __expf(mrow[r] - newm);
      float p0 = __expf(x0 - newm);
      float p1 = __expf(x1 - newm);
      float rs = p0 + p1;
#pragma unroll
      for (int off = 1; off < 16; off <<= 1)
        rs += __shfl_xor(rs, off, 16);
      lrow[r] = lrow[r] * alpha + rs;
      mrow[r] = newm;
#pragma unroll
      for (int f = 0; f < 8; ++f) o[f][r] *= alpha;
      Plds[m * 40 + l16]      = (_Float16)p0;
      Plds[m * 40 + 16 + l16] = (_Float16)p1;
    }
    __syncthreads();
    // re-read P as a 16x32 A-fragment
    v16h pa = cat8(*(const v8h*)&Plds[l16 * 40 + hl * 8],
                   *(const v8h*)&Plds[l16 * 40 + 16 + hl * 8]);
    __syncthreads();
    // O += P * V over 8 d-column tiles
#pragma unroll
    for (int f = 0; f < 8; ++f) {
      const _Float16* vrow = Vb + (size_t)(f * 16 + l16) * T_SEQ + k0 + hl * 8;
      v16h vb = cat8(*(const v8h*)(vrow), *(const v8h*)(vrow + 16));
      o[f] = __builtin_amdgcn_wmma_f32_16x16x32_f16(false, pa, false, vb, (short)0, o[f], false, false);
    }
  }
  // normalize + store ctx as [b, t, h*128+d]
#pragma unroll
  for (int r = 0; r < 8; ++r) {
    int m = r + 8 * hl;
    float inv = 1.0f / lrow[r];
    size_t dst = (size_t)(b * T_SEQ + q0 + m) * D_MODEL + h * HDIM;
#pragma unroll
    for (int f = 0; f < 8; ++f)
      ctx[dst + f * 16 + l16] = (_Float16)(o[f][r] * inv);
  }
}

// ----------------------------------------------------------------------- host
extern "C" void kernel_launch(void* const* d_in, const int* in_sizes, int n_in,
                              void* d_out, int out_size, void* d_ws, size_t ws_size,
                              hipStream_t stream) {
  (void)in_sizes; (void)n_in; (void)out_size; (void)ws_size;
  const float* x     = (const float*)d_in[0];
  const float* Wq    = (const float*)d_in[1];
  const float* Wk    = (const float*)d_in[2];
  const float* Wv    = (const float*)d_in[3];
  const float* Wo    = (const float*)d_in[4];
  const float* bo    = (const float*)d_in[5];
  const float* theta = (const float*)d_in[6];

  const size_t MT   = (size_t)NBATCH * T_SEQ;       // 4096 rows
  const size_t SZ_X = MT * D_MODEL;                 // 8M halves
  const size_t SZ_W = (size_t)D_MODEL * D_MODEL;    // 4M halves

  _Float16* ws   = (_Float16*)d_ws;
  _Float16* Xh   = ws;
  _Float16* Wqh  = Xh + SZ_X;
  _Float16* Wkh  = Wqh + SZ_W;
  _Float16* Wvh  = Wkh + SZ_W;
  _Float16* Woh  = Wvh + SZ_W;
  _Float16* Qtmp = Woh + SZ_W;
  _Float16* Ktmp = Qtmp + SZ_X;
  _Float16* Vtmp = Ktmp + SZ_X;
  _Float16* Qhd  = Vtmp + SZ_X;
  _Float16* Khd  = Qhd + SZ_X;
  _Float16* Vtd  = Khd + SZ_X;
  _Float16* ctxh = Qtmp;  // Qtmp dead after rope_qk; reuse for attention output

  cvt_f32_f16<<<(SZ_X + 255) / 256, 256, 0, stream>>>(x, Xh, (int)SZ_X);
  cvt_f32_f16<<<(SZ_W + 255) / 256, 256, 0, stream>>>(Wq, Wqh, (int)SZ_W);
  cvt_f32_f16<<<(SZ_W + 255) / 256, 256, 0, stream>>>(Wk, Wkh, (int)SZ_W);
  cvt_f32_f16<<<(SZ_W + 255) / 256, 256, 0, stream>>>(Wv, Wvh, (int)SZ_W);
  cvt_f32_f16<<<(SZ_W + 255) / 256, 256, 0, stream>>>(Wo, Woh, (int)SZ_W);

  dim3 gb(32, 4);                                   // 4 waves / block
  dim3 gg(D_MODEL / 128, (unsigned)(MT / 32));
  gemm_nt_f16<<<gg, gb, 0, stream>>>(Xh, Wqh, Qtmp, (int)MT, D_MODEL, D_MODEL);
  gemm_nt_f16<<<gg, gb, 0, stream>>>(Xh, Wkh, Ktmp, (int)MT, D_MODEL, D_MODEL);
  gemm_nt_f16<<<gg, gb, 0, stream>>>(Xh, Wvh, Vtmp, (int)MT, D_MODEL, D_MODEL);

  rope_qk<<<(NBATCH * T_SEQ * NHEADS * 64) / 256, 256, 0, stream>>>(
      Qtmp, Ktmp, theta, Qhd, Khd);
  v_transpose<<<(NBATCH * T_SEQ * NHEADS * HDIM) / 256, 256, 0, stream>>>(Vtmp, Vtd);

  attn_kernel<<<dim3(T_SEQ / 16, NHEADS, NBATCH), 32, 0, stream>>>(Qhd, Khd, Vtd, ctxh);

  gemm_nt_f32_bias<<<gg, gb, 0, stream>>>(ctxh, Woh, bo, (float*)d_out,
                                          (int)MT, D_MODEL, D_MODEL);
}
